// TransformerXL_72816875536741
// MI455X (gfx1250) — matrix operand order repaired and checked by reading, not compiled, not run on hardware
//
#include <hip/hip_runtime.h>

// ---- problem constants ----
#define Bn   4
#define Tn   512
#define Mn   512
#define En   1024
#define Hn   16
#define HDn  64
#define FFn  4096
#define Ln   4
#define Sn   (Mn + Tn)        // 1024
#define QKVF (3 * Hn * HDn)   // 3072
#define SCALEF 0.03125f       // 1/sqrt(E) = 1/32

// ---- types for WMMA ----
typedef __attribute__((ext_vector_type(16))) __bf16 v16bf;
typedef __attribute__((ext_vector_type(8)))  float  v8f;

struct alignas(16) U4 { unsigned int a, b, c, d; };
union FragU { v16bf v; U4 q[2]; };

// ---- CDNA5 async global->LDS copy (probe via __has_builtin) ----
#ifndef __has_builtin
#define __has_builtin(x) 0
#endif
#if __has_builtin(__builtin_amdgcn_global_load_async_to_lds_b128)
#define USE_ASYNC_LDS 1
#else
#define USE_ASYNC_LDS 0
#endif

#define AS3 __attribute__((address_space(3)))
#define AS1 __attribute__((address_space(1)))

#if USE_ASYNC_LDS
// builtin expects pointers to 16-byte int vectors (per clang diagnostic)
typedef int v4i_t __attribute__((vector_size(16)));
__device__ __forceinline__ AS1 v4i_t* to_glb(const void* p) {
  return (AS1 v4i_t*)(unsigned long long)p;
}
__device__ __forceinline__ AS3 v4i_t* to_lds(void* p) {
  // generic LDS address: low 32 bits are the LDS offset (ISA 10.2 aperture rules)
  return (AS3 v4i_t*)(unsigned int)(unsigned long long)p;
}
__device__ __forceinline__ void wait_async0() {
#if __has_builtin(__builtin_amdgcn_s_wait_asynccnt)
  __builtin_amdgcn_s_wait_asynccnt(0);
#else
  asm volatile("s_wait_asynccnt 0x0" ::: "memory");
#endif
}
#endif

// ---- bf16 helpers (RNE) ----
__device__ __forceinline__ unsigned short f2bf(float f) {
  unsigned int u = __float_as_uint(f);
  u += 0x7FFFu + ((u >> 16) & 1u);
  return (unsigned short)(u >> 16);
}
__device__ __forceinline__ float bf2f(unsigned short h) {
  return __uint_as_float(((unsigned int)h) << 16);
}

// =====================================================================
// Generic batched WMMA GEMM:  C[m,n] (+)= sum_k A[m,k] * Bt[n,k]
// A, Bt bf16 (as ushort), C f32. Block tile 64x64, K-step 32, 4 waves.
// Double-buffered LDS pipeline; tiles stream in via async-to-LDS while
// WMMAs consume the previous tile.
// mode 0: store; mode 2: TXL rel-shift scatter-add C[m, n + m - shift] += v
// =====================================================================
__global__ __launch_bounds__(128) void wmma_gemm_bf16(
    const unsigned short* __restrict__ A,
    const unsigned short* __restrict__ Bt,
    float* __restrict__ C,
    int Ndim, int Kdim, int lda, int ldb, int ldc,
    long long aStride, long long bStride, long long cStride,
    int mode, int shiftAmt)
{
  __shared__ unsigned short As[2][64 * 32];
  __shared__ unsigned short Bs[2][64 * 32];

  const int tid  = threadIdx.x;
  const int lane = tid & 31;
  const int wave = tid >> 5;
  const int wr   = (wave >> 1) & 1;   // wave row (0..1) -> 32 rows
  const int wc   = wave & 1;          // wave col (0..1) -> 32 cols
  const int z    = blockIdx.z;

  const unsigned short* Ab = A + (long long)z * aStride;
  const unsigned short* Bb = Bt + (long long)z * bStride;
  float* Cb = C + (long long)z * cStride;

  const int rowBase = blockIdx.y * 64;
  const int colBase = blockIdx.x * 64;

  v8f acc00 = {}; v8f acc01 = {}; v8f acc10 = {}; v8f acc11 = {};

  // issue one 64x32 A tile + one 64x32 B tile into LDS buffer `buf`
  auto issue_tile = [&](int k0, int buf) {
#pragma unroll
    for (int it = 0; it < 2; ++it) {
      int idx = tid + it * 128;
      int row = idx >> 2;
      int seg = idx & 3;
      const unsigned short* ga = Ab + (long long)(rowBase + row) * lda + k0 + seg * 8;
      const unsigned short* gb = Bb + (long long)(colBase + row) * ldb + k0 + seg * 8;
#if USE_ASYNC_LDS
      __builtin_amdgcn_global_load_async_to_lds_b128(
          to_glb(ga), to_lds(&As[buf][row * 32 + seg * 8]), 0, 0);
      __builtin_amdgcn_global_load_async_to_lds_b128(
          to_glb(gb), to_lds(&Bs[buf][row * 32 + seg * 8]), 0, 0);
#else
      *reinterpret_cast<U4*>(&As[buf][row * 32 + seg * 8]) =
          *reinterpret_cast<const U4*>(ga);
      *reinterpret_cast<U4*>(&Bs[buf][row * 32 + seg * 8]) =
          *reinterpret_cast<const U4*>(gb);
      __builtin_prefetch(ga + 32, 0, 1);
      __builtin_prefetch(gb + 32, 0, 1);
#endif
    }
  };

  const int nk = Kdim / 32;
  issue_tile(0, 0);
#if USE_ASYNC_LDS
  wait_async0();
#endif
  __syncthreads();

  for (int kt = 0; kt < nk; ++kt) {
    const int buf = kt & 1;
    if (kt + 1 < nk) issue_tile((kt + 1) * 32, buf ^ 1);

    // fragment assembly per ISA 16-bit A(16x32) / B(32x16) lane layouts
    const int m  = lane & 15;
    const int hk = lane >> 4;
    FragU a0, a1, b0, b1;
    {
      const U4* p0 = reinterpret_cast<const U4*>(&As[buf][(wr * 32 + m) * 32]);
      a0.q[0] = p0[hk];     a0.q[1] = p0[2 + hk];
      const U4* p1 = reinterpret_cast<const U4*>(&As[buf][(wr * 32 + 16 + m) * 32]);
      a1.q[0] = p1[hk];     a1.q[1] = p1[2 + hk];
      const U4* q0 = reinterpret_cast<const U4*>(&Bs[buf][(wc * 32 + m) * 32]);
      b0.q[0] = q0[hk]; b0.q[1] = q0[2 + hk];
      const U4* q1 = reinterpret_cast<const U4*>(&Bs[buf][(wc * 32 + 16 + m) * 32]);
      b1.q[0] = q1[hk]; b1.q[1] = q1[2 + hk];
    }
    acc00 = __builtin_amdgcn_wmma_f32_16x16x32_bf16(false, a0.v, false, b0.v, (short)0, acc00, false, false);
    acc01 = __builtin_amdgcn_wmma_f32_16x16x32_bf16(false, a0.v, false, b1.v, (short)0, acc01, false, false);
    acc10 = __builtin_amdgcn_wmma_f32_16x16x32_bf16(false, a1.v, false, b0.v, (short)0, acc10, false, false);
    acc11 = __builtin_amdgcn_wmma_f32_16x16x32_bf16(false, a1.v, false, b1.v, (short)0, acc11, false, false);

#if USE_ASYNC_LDS
    wait_async0();            // next tile fully landed in LDS
#endif
    __syncthreads();          // everyone done reading `buf`; next tile visible
  }

  // epilogue: C/D layout -> lane holds rows r+8*(lane>>4), col = lane&15
  const int rOff = (lane >> 4) * 8;
  const int cOff = lane & 15;
#pragma unroll
  for (int i = 0; i < 2; ++i) {
#pragma unroll
    for (int j = 0; j < 2; ++j) {
      v8f accv = (i == 0) ? ((j == 0) ? acc00 : acc01) : ((j == 0) ? acc10 : acc11);
      int gr0 = rowBase + wr * 32 + i * 16 + rOff;
      int gc  = colBase + wc * 32 + j * 16 + cOff;
#pragma unroll
      for (int r = 0; r < 8; ++r) {
        int row = gr0 + r;
        float val = accv[r];
        if (mode == 0) {
          Cb[(long long)row * ldc + gc] = val;
        } else { // rel-shift scatter-add: j_out = gc + row - (T-1)
          int jj = gc + row - shiftAmt;
          if (jj >= 0 && jj < Ndim) Cb[(long long)row * ldc + jj] += val;
        }
      }
    }
  }
}

// =====================================================================
// elementwise / reduction kernels
// =====================================================================
__global__ void cvt_f32_bf16_kernel(const float* __restrict__ s,
                                    unsigned short* __restrict__ d, long long n) {
  long long i = (long long)blockIdx.x * blockDim.x + threadIdx.x;
  if (i < n) d[i] = f2bf(s[i]);
}

__global__ void posenc_kernel(unsigned short* __restrict__ pe, long long n) {
  long long i = (long long)blockIdx.x * blockDim.x + threadIdx.x;
  if (i >= n) return;
  int s = (int)(i / En), e = (int)(i % En);
  float p   = (float)(Sn - 1 - s);
  float div = expf(-(logf(10000.0f) / (float)En) * (float)((e >> 1) * 2));
  float ang = p * div;
  pe[i] = f2bf((e & 1) ? cosf(ang) : sinf(ang));
}

__global__ void build_h_kernel(const float* __restrict__ x, float* __restrict__ h, long long n) {
  long long i = (long long)blockIdx.x * blockDim.x + threadIdx.x;
  if (i >= n) return;
  int e = (int)(i % En); long long r = i / En;
  int b = (int)(r % Bn), t = (int)(r / Bn);
  h[i] = x[((long long)b * Tn + t) * En + e];
}

__global__ void build_cat_kernel(const float* __restrict__ mems_i, const float* __restrict__ h,
                                 unsigned short* __restrict__ cat, long long n) {
  long long i = (long long)blockIdx.x * blockDim.x + threadIdx.x;
  if (i >= n) return;
  int e = (int)(i % En); long long r = i / En;
  int b = (int)(r % Bn), s = (int)(r / Bn);
  float v = (s < Mn) ? mems_i[((long long)s * Bn + b) * En + e]
                     : h[((long long)(s - Mn) * Bn + b) * En + e];
  cat[i] = f2bf(v);
}

__global__ void build_quqv_kernel(const float* __restrict__ qkv,
                                  const float* __restrict__ u, const float* __restrict__ v,
                                  unsigned short* __restrict__ qu, unsigned short* __restrict__ qv,
                                  long long n) {
  long long i = (long long)blockIdx.x * blockDim.x + threadIdx.x;
  if (i >= n) return;
  int d = (int)(i % HDn);
  int t = (int)((i / HDn) % Tn);
  int h = (int)((i / ((long long)HDn * Tn)) % Hn);
  int b = (int)(i / ((long long)HDn * Tn * Hn));
  long long qrow = (long long)(Mn + t) * Bn + b;
  float q = qkv[qrow * QKVF + h * HDn + d];
  qu[i] = f2bf(q + u[h * HDn + d]);
  qv[i] = f2bf(q + v[h * HDn + d]);
}

__global__ void build_kvt_kernel(const float* __restrict__ qkv,
                                 unsigned short* __restrict__ kh, unsigned short* __restrict__ vt,
                                 long long n) {
  long long i = (long long)blockIdx.x * blockDim.x + threadIdx.x;
  if (i >= n) return;
  int d = (int)(i % HDn);
  int s = (int)((i / HDn) % Sn);
  int h = (int)((i / ((long long)HDn * Sn)) % Hn);
  int b = (int)(i / ((long long)HDn * Sn * Hn));
  long long srow = (long long)s * Bn + b;
  kh[i] = f2bf(qkv[srow * QKVF + En + h * HDn + d]);
  vt[(((long long)(b * Hn + h) * HDn + d) * Sn) + s] = f2bf(qkv[srow * QKVF + 2 * En + h * HDn + d]);
}

__global__ void rearrange_r_kernel(const float* __restrict__ rf, unsigned short* __restrict__ rh,
                                   long long n) {
  long long i = (long long)blockIdx.x * blockDim.x + threadIdx.x;
  if (i >= n) return;
  int s = (int)(i / En), f = (int)(i % En);
  int h = f / HDn, d = f % HDn;
  rh[((long long)h * Sn + s) * HDn + d] = f2bf(rf[i]);
}

__global__ void gelu_bias_kernel(const float* __restrict__ ff, const float* __restrict__ b1,
                                 unsigned short* __restrict__ out, long long n) {
  long long i = (long long)blockIdx.x * blockDim.x + threadIdx.x;
  if (i >= n) return;
  float x = ff[i] + b1[(int)(i % FFn)];
  out[i] = f2bf(0.5f * x * (1.0f + erff(x * 0.70710678118654752f)));
}

__global__ void final_out_kernel(const float* __restrict__ h, float* __restrict__ out, long long n) {
  long long i = (long long)blockIdx.x * blockDim.x + threadIdx.x;
  if (i >= n) return;
  int e = (int)(i % En); long long r = i / En;
  int t = (int)(r % Tn), b = (int)(r / Tn);
  out[i] = h[((long long)t * Bn + b) * En + e];
}

// softmax over j with causal-mem mask (j <= M+i unmasked), scaled, output bf16
__global__ __launch_bounds__(128) void softmax_kernel(const float* __restrict__ sc,
                                                      unsigned short* __restrict__ attn,
                                                      int memLen, float scale) {
  __shared__ float rowbuf[Sn];
  __shared__ float red[128];
  const int tid = threadIdx.x;
  const long long row = blockIdx.x;          // h*Tn + i
  const int i = (int)(row % Tn);
  const int limit = memLen + i;
  const float* src = sc + row * Sn;
  unsigned short* dst = attn + row * Sn;
  float mx = -3.0e38f;
  for (int j = tid; j < Sn; j += 128) {
    float v = src[j] * scale;
    rowbuf[j] = v;
    if (j <= limit) mx = fmaxf(mx, v);
  }
  red[tid] = mx; __syncthreads();
  for (int s = 64; s > 0; s >>= 1) { if (tid < s) red[tid] = fmaxf(red[tid], red[tid + s]); __syncthreads(); }
  mx = red[0]; __syncthreads();
  float sum = 0.f;
  for (int j = tid; j < Sn; j += 128) if (j <= limit) sum += expf(rowbuf[j] - mx);
  red[tid] = sum; __syncthreads();
  for (int s = 64; s > 0; s >>= 1) { if (tid < s) red[tid] += red[tid + s]; __syncthreads(); }
  float inv = 1.0f / red[0];
  for (int j = tid; j < Sn; j += 128)
    dst[j] = f2bf(j <= limit ? expf(rowbuf[j] - mx) * inv : 0.0f);
}

// out = LayerNorm(x + y (+ bias)) * g + b ; optional bf16 mirror
__global__ __launch_bounds__(256) void add_ln_kernel(const float* __restrict__ x,
                                                     const float* __restrict__ y,
                                                     const float* __restrict__ bias,
                                                     const float* __restrict__ g,
                                                     const float* __restrict__ bb,
                                                     float* __restrict__ outf,
                                                     unsigned short* __restrict__ outh) {
  __shared__ float red[256];
  const int tid = threadIdx.x;
  const long long row = blockIdx.x;
  const float* xp = x + row * En;
  const float* yp = y + row * En;
  float v[En / 256];
  float s = 0.f;
#pragma unroll
  for (int k = 0; k < En / 256; ++k) {
    int j = tid + k * 256;
    float t = xp[j] + yp[j] + (bias ? bias[j] : 0.f);
    v[k] = t; s += t;
  }
  red[tid] = s; __syncthreads();
  for (int st = 128; st > 0; st >>= 1) { if (tid < st) red[tid] += red[tid + st]; __syncthreads(); }
  float mean = red[0] * (1.0f / En); __syncthreads();
  float ss = 0.f;
#pragma unroll
  for (int k = 0; k < En / 256; ++k) { float d = v[k] - mean; ss += d * d; }
  red[tid] = ss; __syncthreads();
  for (int st = 128; st > 0; st >>= 1) { if (tid < st) red[tid] += red[tid + st]; __syncthreads(); }
  float rstd = rsqrtf(red[0] * (1.0f / En) + 1e-5f);
#pragma unroll
  for (int k = 0; k < En / 256; ++k) {
    int j = tid + k * 256;
    float o = (v[k] - mean) * rstd * g[j] + bb[j];
    outf[row * En + j] = o;
    if (outh) outh[row * En + j] = f2bf(o);
  }
}

// =====================================================================
// host orchestration
// =====================================================================
extern "C" void kernel_launch(void* const* d_in, const int* in_sizes, int n_in,
                              void* d_out, int out_size, void* d_ws, size_t ws_size,
                              hipStream_t stream) {
  (void)in_sizes; (void)n_in; (void)out_size; (void)ws_size;

  const float* x      = (const float*)d_in[0];
  // d_in[1] positions, d_in[2] attn_mask: reproduced analytically
  const float* mems   = (const float*)d_in[3];
  const float* u_bias = (const float*)d_in[4];
  const float* v_bias = (const float*)d_in[5];
  const float* W_qkv  = (const float*)d_in[6];
  const float* W_pos  = (const float*)d_in[7];
  const float* W_out  = (const float*)d_in[8];
  const float* W1     = (const float*)d_in[9];
  const float* b1     = (const float*)d_in[10];
  const float* W2     = (const float*)d_in[11];
  const float* b2     = (const float*)d_in[12];
  const float* g1     = (const float*)d_in[13];
  const float* be1    = (const float*)d_in[14];
  const float* g2     = (const float*)d_in[15];
  const float* be2    = (const float*)d_in[16];

  char* ws = (char*)d_ws;
  size_t off = 0;
  auto alloc = [&](size_t bytes) -> void* {
    void* p = ws + off;
    off = (off + bytes + 255) & ~(size_t)255;
    return p;
  };
  typedef unsigned short us;
  us*  wqkv_h = (us*)alloc((size_t)QKVF * En * 2);
  us*  wpos_h = (us*)alloc((size_t)En * En * 2);
  us*  wout_h = (us*)alloc((size_t)En * En * 2);
  us*  w1_h   = (us*)alloc((size_t)FFn * En * 2);
  us*  w2_h   = (us*)alloc((size_t)En * FFn * 2);
  us*  pe_h   = (us*)alloc((size_t)Sn * En * 2);
  float* h_f  = (float*)alloc((size_t)Tn * Bn * En * 4);
  us*  cat_h  = (us*)alloc((size_t)Sn * Bn * En * 2);
  float* qkv_f = (float*)alloc((size_t)Sn * Bn * QKVF * 4);
  float* r_f   = (float*)alloc((size_t)Sn * En * 4);
  us*  r_h    = (us*)alloc((size_t)Sn * En * 2);
  us*  qu_h   = (us*)alloc((size_t)Bn * Hn * Tn * HDn * 2);
  us*  qv_h   = (us*)alloc((size_t)Bn * Hn * Tn * HDn * 2);
  us*  k_h    = (us*)alloc((size_t)Bn * Hn * Sn * HDn * 2);
  us*  vt_h   = (us*)alloc((size_t)Bn * Hn * HDn * Sn * 2);
  float* sc_f = (float*)alloc((size_t)Hn * Tn * Sn * 4);   // per-b reuse
  us*  at_h   = (us*)alloc((size_t)Hn * Tn * Sn * 2);      // per-b reuse
  float* ctx_f = (float*)alloc((size_t)Tn * Bn * En * 4);
  us*  ctx_h  = (us*)alloc((size_t)Tn * Bn * En * 2);
  float* ao_f  = (float*)alloc((size_t)Tn * Bn * En * 4);
  float* o1_f  = (float*)alloc((size_t)Tn * Bn * En * 4);
  us*  o1_h   = (us*)alloc((size_t)Tn * Bn * En * 2);
  float* ff1_f = (float*)alloc((size_t)Tn * Bn * FFn * 4);
  us*  ff1_h  = (us*)alloc((size_t)Tn * Bn * FFn * 2);
  float* ff2_f = (float*)alloc((size_t)Tn * Bn * En * 4);

  auto ew = [&](long long n) { return dim3((unsigned)((n + 255) / 256)); };
  auto gemm = [&](const us* A, const us* Bt, float* C, int Mdim, int Ndim, int Kdim,
                  int lda, int ldb, int ldc, long long aS, long long bS, long long cS,
                  int batches, int mode, int shiftAmt) {
    dim3 grid((unsigned)(Ndim / 64), (unsigned)(Mdim / 64), (unsigned)batches);
    wmma_gemm_bf16<<<grid, 128, 0, stream>>>(A, Bt, C, Ndim, Kdim, lda, ldb, ldc,
                                             aS, bS, cS, mode, shiftAmt);
  };

  // static prep
  {
    long long n = (long long)Sn * En;
    posenc_kernel<<<ew(n), 256, 0, stream>>>(pe_h, n);
    n = (long long)Tn * Bn * En;
    build_h_kernel<<<ew(n), 256, 0, stream>>>(x, h_f, n);
  }

  for (int i = 0; i < Ln; ++i) {
    // per-layer weight conversion f32 -> bf16
    long long n;
    n = (long long)QKVF * En; cvt_f32_bf16_kernel<<<ew(n), 256, 0, stream>>>(W_qkv + (size_t)i * n, wqkv_h, n);
    n = (long long)En * En;   cvt_f32_bf16_kernel<<<ew(n), 256, 0, stream>>>(W_pos + (size_t)i * n, wpos_h, n);
    n = (long long)En * En;   cvt_f32_bf16_kernel<<<ew(n), 256, 0, stream>>>(W_out + (size_t)i * n, wout_h, n);
    n = (long long)FFn * En;  cvt_f32_bf16_kernel<<<ew(n), 256, 0, stream>>>(W1 + (size_t)i * n, w1_h, n);
    n = (long long)En * FFn;  cvt_f32_bf16_kernel<<<ew(n), 256, 0, stream>>>(W2 + (size_t)i * n, w2_h, n);

    // cat = [mems_i ; h]  (S,B,E) bf16
    n = (long long)Sn * Bn * En;
    build_cat_kernel<<<ew(n), 256, 0, stream>>>(mems + (size_t)i * Mn * Bn * En, h_f, cat_h, n);

    // qkv = cat @ Wqkv^T   (4096 x 3072)
    gemm(cat_h, wqkv_h, qkv_f, Sn * Bn, QKVF, En, En, En, QKVF, 0, 0, 0, 1, 0, 0);
    // r = pos_enc @ Wpos^T (1024 x 1024)
    gemm(pe_h, wpos_h, r_f, Sn, En, En, En, En, En, 0, 0, 0, 1, 0, 0);
    n = (long long)Sn * En;
    rearrange_r_kernel<<<ew(n), 256, 0, stream>>>(r_f, r_h, n);

    n = (long long)Bn * Hn * Tn * HDn;
    build_quqv_kernel<<<ew(n), 256, 0, stream>>>(qkv_f, u_bias, v_bias, qu_h, qv_h, n);
    n = (long long)Bn * Hn * Sn * HDn;
    build_kvt_kernel<<<ew(n), 256, 0, stream>>>(qkv_f, k_h, vt_h, n);

    // attention, per batch index b (16 heads batched in grid.z)
    for (int b = 0; b < Bn; ++b) {
      const us* quB = qu_h + (size_t)b * Hn * Tn * HDn;
      const us* qvB = qv_h + (size_t)b * Hn * Tn * HDn;
      const us* kB  = k_h + (size_t)b * Hn * Sn * HDn;
      const us* vtB = vt_h + (size_t)b * Hn * HDn * Sn;
      // content: scores[h,i,j] = (q+u)·k
      gemm(quB, kB, sc_f, Tn, Sn, HDn, HDn, HDn, Sn,
           (long long)Tn * HDn, (long long)Sn * HDn, (long long)Tn * Sn, Hn, 0, 0);
      // pos: raw[i,k] = (q+v)·r_k scattered to scores[i, k + i - (T-1)] (rel-shift)
      gemm(qvB, r_h, sc_f, Tn, Sn, HDn, HDn, HDn, Sn,
           (long long)Tn * HDn, (long long)Sn * HDn, (long long)Tn * Sn, Hn, 2, Tn - 1);
      // masked softmax -> bf16 probs
      softmax_kernel<<<dim3(Hn * Tn), 128, 0, stream>>>(sc_f, at_h, Mn, SCALEF);
      // ctx[i,d] = sum_j attn[i,j] * v[j,d]  (Bt = v^T), write into (T,B,E) layout
      gemm(at_h, vtB, ctx_f + (size_t)b * En, Tn, HDn, Sn, Sn, Sn, Bn * En,
           (long long)Tn * Sn, (long long)HDn * Sn, (long long)HDn, Hn, 0, 0);
    }

    n = (long long)Tn * Bn * En;
    cvt_f32_bf16_kernel<<<ew(n), 256, 0, stream>>>(ctx_f, ctx_h, n);
    // attn_out = ctx @ Wout^T
    gemm(ctx_h, wout_h, ao_f, Tn * Bn, En, En, En, En, En, 0, 0, 0, 1, 0, 0);
    // out1 = LN(h + attn_out)
    add_ln_kernel<<<dim3(Tn * Bn), 256, 0, stream>>>(h_f, ao_f, nullptr,
        g1 + (size_t)i * En, be1 + (size_t)i * En, o1_f, o1_h);
    // ff1 = out1 @ W1^T ; gelu(ff1 + b1)
    gemm(o1_h, w1_h, ff1_f, Tn * Bn, FFn, En, En, En, FFn, 0, 0, 0, 1, 0, 0);
    n = (long long)Tn * Bn * FFn;
    gelu_bias_kernel<<<ew(n), 256, 0, stream>>>(ff1_f, b1 + (size_t)i * FFn, ff1_h, n);
    // ff2 = gelu @ W2^T ; h = LN(out1 + ff2 + b2)
    gemm(ff1_h, w2_h, ff2_f, Tn * Bn, En, FFn, FFn, FFn, En, 0, 0, 0, 1, 0, 0);
    add_ln_kernel<<<dim3(Tn * Bn), 256, 0, stream>>>(o1_f, ff2_f, b2 + (size_t)i * En,
        g2 + (size_t)i * En, be2 + (size_t)i * En, h_f, nullptr);
  }

  long long n = (long long)Bn * Tn * En;
  final_out_kernel<<<ew(n), 256, 0, stream>>>(h_f, (float*)d_out, n);
}
